// LIIFquery_19275813224620
// MI455X (gfx1250) — matrix operand order repaired and checked by reading, not compiled, not run on hardware
//
#include <hip/hip_runtime.h>

// ---------------------------------------------------------------------------
// LIIF query for MI455X (gfx1250): bf16 WMMA MLP, L2-resident data.
// 512 threads / 16 waves per block, one N-tile per wave (keeps VGPRs < 256).
// ---------------------------------------------------------------------------

typedef __attribute__((ext_vector_type(16))) __bf16 v16bf;
typedef __attribute__((ext_vector_type(8)))  float  v8f;

#define HH   48
#define WW   48
#define HW   2304          // 48*48
#define CF   512           // feature channels
#define HID  256
#define QN   65536
#define BN   2
#define K0   544           // padded layer-0 K (516 -> 17*32)
#define LDA  552           // LDS stride (halves) for the wide activation buffer
#define LDB  264           // LDS stride (halves) for the 256-wide buffer
#define TM   64            // MLP points per block (16 queries * 4 shifts)
#define NQ   16            // queries per block
#define NTHREADS 512       // 16 waves

union FragU { uint4 q[2]; v16bf v; };

__device__ __forceinline__ unsigned short f2bf(float f) {
  union { float f; unsigned u; } c; c.f = f;
  unsigned u = c.u;
  return (unsigned short)((u + 0x7FFFu + ((u >> 16) & 1u)) >> 16); // RNE
}

__device__ __forceinline__ int imin(int a, int b) { return a < b ? a : b; }
__device__ __forceinline__ int imax(int a, int b) { return a > b ? a : b; }

// A-matrix 16x32 bf16 fragment (ISA 7.12.2): M = lane&15,
// halves [0..7] = K (lane>>4)*8 .. +7 ; halves [8..15] = +16.
__device__ __forceinline__ v16bf load_fragA(const unsigned short* m, int ld,
                                            int rowbase, int kk, int lane) {
  const uint4* p = (const uint4*)(m + (size_t)(rowbase + (lane & 15)) * (size_t)ld)
                   + (kk * 4 + (lane >> 4));
  FragU u; u.q[0] = p[0]; u.q[1] = p[2];
  return u.v;
}

// B-matrix 32x16 bf16 fragment, sourced from W^T rows: N = lane&15,
// halves [0..7] = K (lane>>4)*8 .. +7 ; halves [8..15] = +16.
__device__ __forceinline__ v16bf load_fragB(const unsigned short* m, int ld,
                                            int rowbase, int kk, int lane) {
  const uint4* p = (const uint4*)(m + (size_t)(rowbase + (lane & 15)) * (size_t)ld)
                   + (kk * 4 + (lane >> 4));
  FragU u; u.q[0] = p[0]; u.q[1] = p[2];
  return u.v;
}

// One hidden layer: 64 points x 256 outputs. Wave `wave` owns N-tile nt=wave.
// Weight fragments are loop-invariant across mt -> compiler keeps them in VGPRs.
template <int KSTEPS>
__device__ __forceinline__ void mlp_hidden(const unsigned short* __restrict__ Wt, int ldw,
                                           const float* __restrict__ bias,
                                           const unsigned short* inB, int ldi,
                                           unsigned short* outB, int ldo,
                                           int wave, int lane) {
  const int nt = wave;
  const int n = lane & 15;
  const float bv = bias[nt * 16 + n];
  for (int mt = 0; mt < 4; ++mt) {
    v8f acc = {0.f, 0.f, 0.f, 0.f, 0.f, 0.f, 0.f, 0.f};
#pragma unroll
    for (int kk = 0; kk < KSTEPS; ++kk) {
      v16bf va = load_fragA(inB, ldi, mt * 16, kk, lane);
      v16bf vb = load_fragB(Wt, ldw, nt * 16, kk, lane);
      acc = __builtin_amdgcn_wmma_f32_16x16x32_bf16(false, va, false, vb,
                                                    (short)0, acc, false, false);
    }
    // C/D layout: VGPR r, lanes 0-15 -> M=r ; lanes 16-31 -> M=r+8 ; N=lane&15
    const int mbase = mt * 16 + ((lane >> 4) << 3);
#pragma unroll
    for (int r = 0; r < 8; ++r) {
      float v = fmaxf(acc[r] + bv, 0.f);
      outB[(size_t)(mbase + r) * ldo + nt * 16 + n] = f2bf(v);
    }
  }
}

// Output layer: 64 points x 16 (3 real) outputs, f32 result to LDS.
__device__ __forceinline__ void mlp_out(const unsigned short* __restrict__ Wt, int ldw,
                                        const float* __restrict__ b4,
                                        const unsigned short* inB, int ldi,
                                        float* pred, int wave, int lane) {
  const int mt = wave;
  v8f acc = {0.f, 0.f, 0.f, 0.f, 0.f, 0.f, 0.f, 0.f};
#pragma unroll
  for (int kk = 0; kk < 8; ++kk) {
    v16bf va = load_fragA(inB, ldi, mt * 16, kk, lane);
    v16bf vb = load_fragB(Wt, ldw, 0, kk, lane);
    acc = __builtin_amdgcn_wmma_f32_16x16x32_bf16(false, va, false, vb,
                                                  (short)0, acc, false, false);
  }
  const int n = lane & 15;
  const float bv = (n < 3) ? b4[n] : 0.f;
  const int mbase = mt * 16 + ((lane >> 4) << 3);
#pragma unroll
  for (int r = 0; r < 8; ++r) pred[(mbase + r) * 16 + n] = acc[r] + bv;
}

// --------------------------- prep kernels ----------------------------------

// feat [B,C,H,W] f32  ->  featT [B, H*W, C] bf16 (contiguous gather rows)
__global__ void prep_feat(const float* __restrict__ feat,
                          unsigned short* __restrict__ featT) {
  int i = blockIdx.x * blockDim.x + threadIdx.x;
  if (i >= BN * HW * CF) return;
  int c    = i & (CF - 1);
  int rest = i >> 9;               // b*HW + hw
  int b    = rest / HW;
  int hw   = rest - b * HW;
  featT[i] = f2bf(feat[((size_t)b * CF + c) * HW + hw]);
}

// Weights f32 [K][N] -> bf16 transposed [N][K], zero-padded.
__global__ void prep_weights(const float* __restrict__ W0, const float* __restrict__ W1,
                             const float* __restrict__ W2, const float* __restrict__ W3,
                             const float* __restrict__ W4,
                             unsigned short* __restrict__ W0t, unsigned short* __restrict__ W1t,
                             unsigned short* __restrict__ W2t, unsigned short* __restrict__ W3t,
                             unsigned short* __restrict__ W4t) {
  int i = blockIdx.x * blockDim.x + threadIdx.x;
  if (i < 256 * K0) {
    int nn = i / K0, k = i - nn * K0;
    W0t[i] = (k < 516) ? f2bf(W0[k * 256 + nn]) : (unsigned short)0;
  }
  if (i < 256 * 256) {
    int nn = i >> 8, k = i & 255;
    W1t[i] = f2bf(W1[k * 256 + nn]);
    W2t[i] = f2bf(W2[k * 256 + nn]);
    W3t[i] = f2bf(W3[k * 256 + nn]);
  }
  if (i < 16 * 256) {
    int nn = i >> 8, k = i & 255;
    W4t[i] = (nn < 3) ? f2bf(W4[k * 3 + nn]) : (unsigned short)0;
  }
}

// ----------------------------- main kernel ---------------------------------

__global__ __launch_bounds__(NTHREADS) void liif_main(
    const float* __restrict__ coord, const float* __restrict__ cell,
    const unsigned short* __restrict__ featT,
    const unsigned short* __restrict__ W0t, const float* __restrict__ b0,
    const unsigned short* __restrict__ W1t, const float* __restrict__ b1,
    const unsigned short* __restrict__ W2t, const float* __restrict__ b2,
    const unsigned short* __restrict__ W3t, const float* __restrict__ b3,
    const unsigned short* __restrict__ W4t, const float* __restrict__ b4,
    float* __restrict__ out) {
  extern __shared__ unsigned char smem[];
  unsigned short* bufA = (unsigned short*)smem;                       // [TM][LDA]
  unsigned short* bufB = (unsigned short*)(smem + TM * LDA * 2);      // [TM][LDB]
  float* pred  = (float*)bufB;                                        // overlap (layer4)
  float* sArea = (float*)(smem + TM * LDA * 2 + TM * LDB * 2);        // [TM]
  int*   sLin  = (int*)(sArea + TM);                                  // [TM]

  const int t    = threadIdx.x;
  const int lane = t & 31;
  const int wave = t >> 5;

  // ---- stage 1: per-point scalars (nearest idx, rel coord, area) ----
  if (t < TM) {
    int qi = t >> 2, s = t & 3;
    int gq = blockIdx.x * NQ + qi;          // global query = b*QN + q
    int b  = gq >> 16;                      // QN == 65536
    float c0  = coord[(size_t)gq * 2 + 0];
    float c1  = coord[(size_t)gq * 2 + 1];
    float ce0 = cell[(size_t)gq * 2 + 0];
    float ce1 = cell[(size_t)gq * 2 + 1];
    float vx = (s & 2) ? 1.f : -1.f;        // shifts [(-1,-1),(-1,1),(1,-1),(1,1)]
    float vy = (s & 1) ? 1.f : -1.f;
    const float r48 = 1.f / 48.f;
    float cx = fminf(fmaxf(c0 + vx * r48 + 1e-6f, -1.f + 1e-6f), 1.f - 1e-6f);
    float cy = fminf(fmaxf(c1 + vy * r48 + 1e-6f, -1.f + 1e-6f), 1.f - 1e-6f);
    int ir = imin(imax((int)rintf(((cx + 1.f) * 48.f - 1.f) * 0.5f), 0), HH - 1);
    int ic = imin(imax((int)rintf(((cy + 1.f) * 48.f - 1.f) * 0.5f), 0), WW - 1);
    sLin[t] = b * HW + ir * WW + ic;
    float qr = ((float)ir + 0.5f) * (2.f / 48.f) - 1.f;
    float qc = ((float)ic + 0.5f) * (2.f / 48.f) - 1.f;
    float r0 = (c0 - qr) * 48.f;
    float r1 = (c1 - qc) * 48.f;
    sArea[t] = fabsf(r0 * r1) + 1e-9f;
    unsigned short* row = bufA + t * LDA;
    row[512] = f2bf(r0);
    row[513] = f2bf(r1);
    row[514] = f2bf(ce0 * 48.f);
    row[515] = f2bf(ce1 * 48.f);
#pragma unroll
    for (int k = 516; k < K0; ++k) row[k] = 0;   // zero pad to K0
  }
  __syncthreads();

  // ---- stage 2: gather 512 bf16 feats/point (contiguous 1KB rows) ----
  {
    int p = t >> 3, sub = t & 7;                  // 8 threads per point
    const uint4* src = (const uint4*)(featT + (size_t)sLin[p] * CF);
    uint4* dst = (uint4*)(bufA + p * LDA);
#pragma unroll
    for (int i = 0; i < 8; ++i) dst[sub + 8 * i] = src[sub + 8 * i];
  }
  __syncthreads();

  // ---- MLP: bufA ->(L0) bufB ->(L1) bufA ->(L2) bufB ->(L3) bufA ->(L4) pred ----
  mlp_hidden<17>(W0t, K0,  b0, bufA, LDA, bufB, LDB, wave, lane);
  __syncthreads();
  mlp_hidden<8>(W1t, HID, b1, bufB, LDB, bufA, LDA, wave, lane);
  __syncthreads();
  mlp_hidden<8>(W2t, HID, b2, bufA, LDA, bufB, LDB, wave, lane);
  __syncthreads();
  mlp_hidden<8>(W3t, HID, b3, bufB, LDB, bufA, LDA, wave, lane);
  __syncthreads();
  if (wave < 4) mlp_out(W4t, HID, b4, bufA, LDA, pred, wave, lane);
  __syncthreads();

  // ---- combine: out = sum_s pred[s] * area[3-s] / tot ----
  if (t < NQ * 3) {
    int qi = t / 3, o = t - qi * 3;
    int p = qi * 4;
    float a0 = sArea[p + 0], a1 = sArea[p + 1], a2 = sArea[p + 2], a3 = sArea[p + 3];
    float tot = a0 + a1 + a2 + a3;
    float acc = pred[(p + 0) * 16 + o] * a3 + pred[(p + 1) * 16 + o] * a2 +
                pred[(p + 2) * 16 + o] * a1 + pred[(p + 3) * 16 + o] * a0;
    int gq = blockIdx.x * NQ + qi;
    out[(size_t)gq * 3 + o] = acc / tot;
  }
}

// ------------------------------ launcher -----------------------------------

extern "C" void kernel_launch(void* const* d_in, const int* in_sizes, int n_in,
                              void* d_out, int out_size, void* d_ws, size_t ws_size,
                              hipStream_t stream) {
  const float* feat  = (const float*)d_in[0];
  const float* coord = (const float*)d_in[1];
  const float* cell  = (const float*)d_in[2];
  const float* W0 = (const float*)d_in[3];
  const float* b0 = (const float*)d_in[4];
  const float* W1 = (const float*)d_in[5];
  const float* b1 = (const float*)d_in[6];
  const float* W2 = (const float*)d_in[7];
  const float* b2 = (const float*)d_in[8];
  const float* W3 = (const float*)d_in[9];
  const float* b3 = (const float*)d_in[10];
  const float* W4 = (const float*)d_in[11];
  const float* b4 = (const float*)d_in[12];
  float* out = (float*)d_out;

  // workspace carve (bf16 halves), all regions 16B aligned
  unsigned short* featT = (unsigned short*)d_ws;            // 2*2304*512
  unsigned short* W0t = featT + (size_t)BN * HW * CF;       // 256*544
  unsigned short* W1t = W0t + 256 * K0;                     // 256*256
  unsigned short* W2t = W1t + 256 * 256;
  unsigned short* W3t = W2t + 256 * 256;
  unsigned short* W4t = W3t + 256 * 256;                    // 16*256

  prep_feat<<<(BN * HW * CF + 255) / 256, 256, 0, stream>>>(feat, featT);
  prep_weights<<<(256 * K0 + 255) / 256, 256, 0, stream>>>(W0, W1, W2, W3, W4,
                                                           W0t, W1t, W2t, W3t, W4t);

  const size_t shmem = (size_t)TM * LDA * 2 + (size_t)TM * LDB * 2 +
                       (size_t)TM * 4 + (size_t)TM * 4;      // ~105 KB
  liif_main<<<(BN * QN) / NQ, NTHREADS, shmem, stream>>>(
      coord, cell, featT, W0t, b0, W1t, b1, W2t, b2, W3t, b3, W4t, b4, out);
}
